// SparseConv3D_75531294867875
// MI455X (gfx1250) — matrix operand order repaired and checked by reading, not compiled, not run on hardware
//
#include <hip/hip_runtime.h>

typedef __attribute__((ext_vector_type(16))) _Float16 v16h;
typedef __attribute__((ext_vector_type(8)))  float    v8f;

#define N_VOX  400000
#define CCH    64
#define DGRID  128
#define WAVES_PER_BLOCK 16
#define BLOCK  (WAVES_PER_BLOCK * 32)
// 27 offsets * 8 fragments (4 n-tiles x 2 k-halves) * 32 lanes * 16 halfs * 2B
#define LDS_BYTES (27 * 8 * 32 * 16 * 2)

union AFrag {
    v16h v;
    _Float16 h[16];
    unsigned u[8];
};

union HPack {
    _Float16 h[2];
    unsigned u;
};

__global__ __launch_bounds__(BLOCK) void sparse_conv3d_wmma_f16(
    const int*   __restrict__ vidx,   // [N,3]
    const float* __restrict__ feat,   // [N,64]
    const float* __restrict__ wgt,    // [27,64,64] (3,3,3,64,64 flattened)
    const float* __restrict__ occ,    // [D,D,D]
    float*       __restrict__ out)    // [N,64]
{
    extern __shared__ _Float16 ldsW[]; // [frag(216)][lane(32)][e(16)]

    const int tid = threadIdx.x;

    // ---- Stage weights as f16 in exact WMMA-B fragment layout -------------
    // fragment f = o*8 + j*2 + h  (j: 16-col tile of C_OUT, h: K-half of C_IN)
    // B element: lane l -> n = l&15, kHi = l>>4 ; element e -> k = kHi*8 + e (e<8), 16 + kHi*8 + (e-8) (e>=8)
    // Two consecutive elements (e, e+1) per iteration -> packed b32 LDS store.
    {
        unsigned* ldsW32 = (unsigned*)ldsW;
        for (int t2 = tid; t2 < (27 * 64 * 64) / 2; t2 += BLOCK) {
            int t    = t2 << 1;
            int f    = t >> 9;        // /512
            int r    = t & 511;
            int lane = r >> 4;
            int e    = r & 15;        // even
            int o    = f >> 3;
            int j    = (f >> 1) & 3;
            int h    = f & 1;
            int cin  = h * 32 + ((e & 8) << 1) + (lane >> 4) * 8 + (e & 7);
            int cout = j * 16 + (lane & 15);
            const float* src = wgt + (o * 64 + cin) * 64 + cout;
            HPack p;
            p.h[0] = (_Float16)src[0];
            p.h[1] = (_Float16)src[64];   // cin+1 -> +64 floats
            ldsW32[t2] = p.u;
        }
    }
    __syncthreads();

    const int wave = tid >> 5;
    const int lane = tid & 31;
    const int m    = lane & 15;   // matrix row owned by this lane
    const int hi   = lane >> 4;
    const int tile = blockIdx.x * WAVES_PER_BLOCK + wave;
    const int row0 = tile * 16;
    if (row0 >= N_VOX) return;    // wave-uniform: EXEC stays all-1 for WMMA
    const int row = row0 + m;

    // ---- 27-bit neighbor occupancy mask for this lane's voxel -------------
    const int ix = vidx[row * 3 + 0];
    const int iy = vidx[row * 3 + 1];
    const int iz = vidx[row * 3 + 2];
    unsigned mask27 = 0;
    int ob = 0;
#pragma unroll
    for (int dx = -1; dx <= 1; ++dx)
#pragma unroll
        for (int dy = -1; dy <= 1; ++dy)
#pragma unroll
            for (int dz = -1; dz <= 1; ++dz) {
                float g = occ[((ix + dx) * DGRID + (iy + dy)) * DGRID + (iz + dz)];
                mask27 |= (g != 0.0f ? 1u : 0u) << ob;
                ++ob;
            }

    // ---- Load + convert A fragments (16 voxels x 64ch -> two 16x32 f16) ---
    // element e -> K = base + hi*8 + e (e<8), base+16+hi*8+(e-8) (e>=8)
    AFrag a0, a1;
    const float* xr = feat + (long)row * CCH;
#pragma unroll
    for (int e = 0; e < 8; ++e) {
        a0.h[e]     = (_Float16)xr[hi * 8 + e];
        a0.h[8 + e] = (_Float16)xr[16 + hi * 8 + e];
        a1.h[e]     = (_Float16)xr[32 + hi * 8 + e];
        a1.h[8 + e] = (_Float16)xr[48 + hi * 8 + e];
    }

    // Split accumulators: one per (n-tile j, K-half). Each is written exactly
    // once per offset -> 8 independent WMMA dependency chains.
    v8f acc0[4], acc1[4];
#pragma unroll
    for (int j = 0; j < 4; ++j) { acc0[j] = (v8f){}; acc1[j] = (v8f){}; }

    // ---- 27 offsets: mask A rows per-lane, accumulate A_m @ W[o] ----------
    // Rolled (unroll 1): keeps live set to one iteration so the allocator can
    // coalesce each accumulator across the backedge (no v_mov shuffles).
#pragma unroll 1
    for (int o = 0; o < 27; ++o) {
        const unsigned sel = ((mask27 >> o) & 1u) ? 0xffffffffu : 0u;
        AFrag am0, am1;
#pragma unroll
        for (int w = 0; w < 8; ++w) {
            am0.u[w] = a0.u[w] & sel;
            am1.u[w] = a1.u[w] & sel;
        }
        const _Float16* bbase = ldsW + (o * 8 << 9) + (lane << 4);
#pragma unroll
        for (int j = 0; j < 4; ++j) {
            v16h b0 = *(const v16h*)(bbase + ((j * 2 + 0) << 9));
            v16h b1 = *(const v16h*)(bbase + ((j * 2 + 1) << 9));
            acc0[j] = __builtin_amdgcn_wmma_f32_16x16x32_f16(
                false, am0.v, false, b0, (short)0, acc0[j], false, false);
            acc1[j] = __builtin_amdgcn_wmma_f32_16x16x32_f16(
                false, am1.v, false, b1, (short)0, acc1[j], false, false);
        }
    }

    // ---- ReLU + store: D vgpr r -> row (hi*8 + r), col = j*16 + (lane&15) -
#pragma unroll
    for (int j = 0; j < 4; ++j) {
        v8f c = acc0[j] + acc1[j];
#pragma unroll
        for (int r = 0; r < 8; ++r) {
            float v = c[r];
            v = v > 0.0f ? v : 0.0f;
            out[(long)(row0 + hi * 8 + r) * CCH + j * 16 + m] = v;
        }
    }
}

extern "C" void kernel_launch(void* const* d_in, const int* in_sizes, int n_in,
                              void* d_out, int out_size, void* d_ws, size_t ws_size,
                              hipStream_t stream) {
    const int*   vidx = (const int*)  d_in[0];   // voxel_idx         [N,3] i32
    const float* feat = (const float*)d_in[1];   // voxel_features    [N,64] f32
    const float* wgt  = (const float*)d_in[2];   // weights           [27,64,64] f32
    const float* occ  = (const float*)d_in[3];   // pts_per_voxel_inv [D,D,D] f32
    float*       out  = (float*)d_out;

    const int tiles  = (N_VOX + 15) / 16;
    const int blocks = (tiles + WAVES_PER_BLOCK - 1) / WAVES_PER_BLOCK;
    sparse_conv3d_wmma_f16<<<blocks, BLOCK, LDS_BYTES, stream>>>(vidx, feat, wgt, occ, out);
}
